// GCNIIBackbone_42004780155161
// MI455X (gfx1250) — matrix operand order repaired and checked by reading, not compiled, not run on hardware
//
#include <hip/hip_runtime.h>
#include <hip/hip_bf16.h>
#include <math.h>

typedef __attribute__((ext_vector_type(16))) _Float16 v16h;
typedef __attribute__((ext_vector_type(8)))  _Float16 v8h;
typedef __attribute__((ext_vector_type(8)))  float    v8f;

#define NN      50000
#define NE      500000
#define DIN     64
#define DHID    128
#define DOUT    64
#define NLAYERS 8
#define LN_EPS  1e-5f

// LDS tile stride (halves): 128+8 pad; 136*2 = 272 B = 17*16 -> rows stay
// 16-byte aligned for b128 LDS fragment loads, lanes spread across banks.
#define STR 136

// ---------------------------------------------------------------------------
// WMMA fragment builders (wave32, v_wmma_f32_16x16x32_f16)
// A (16x32 f16): lane<16 -> M=lane, K=kb+e;
// lane>=16 -> M=lane-16, K=kb+16+e (matches B). 16 K-contiguous halves
// -> two 16B LDS loads.
// B (32x16 f16), contiguous K order (same as A): lane<16 -> N=lane, K=kb+e;
// lane>=16 -> N=lane-16, K=kb+16+e. 16 K-contiguous halves -> one 32B load
// from the PRE-TRANSPOSED weight Wt[n][k].
// ---------------------------------------------------------------------------
__device__ __forceinline__ v16h frag_a_lds(const _Float16* As, int lane,
                                           int kbase) {
  const int m   = lane & 15;
  const int hi8 = (lane & 16) ? 16 : 0;
  const _Float16* p = As + m * STR + kbase + hi8;
  v8h lo = *(const v8h*)(p);
  v8h hi = *(const v8h*)(p + 8);
  return __builtin_shufflevector(lo, hi, 0, 1, 2, 3, 4, 5, 6, 7, 8, 9, 10, 11,
                                 12, 13, 14, 15);
}

__device__ __forceinline__ v16h frag_b_glbT(const _Float16* __restrict__ Wt,
                                            int Kdim, int lane, int kbase,
                                            int n0) {
  const int n    = n0 + (lane & 15);
  const int hi16 = (lane & 16) ? 16 : 0;
  return *(const v16h*)(Wt + (size_t)n * Kdim + kbase + hi16);
}

// ---------------------------------------------------------------------------
// Degree / normalization
// ---------------------------------------------------------------------------
__global__ void deg_init_kernel(float* __restrict__ deg) {
  int i = blockIdx.x * blockDim.x + threadIdx.x;
  if (i < NN) deg[i] = 1.0f;  // self-loop contributes 1 to every node
}

__global__ void deg_count_kernel(const int* __restrict__ cols,
                                 float* __restrict__ deg) {
  int e = blockIdx.x * blockDim.x + threadIdx.x;
  if (e < NE) atomicAdd(&deg[cols[e]], 1.0f);
}

__global__ void deg_to_dinv_kernel(float* __restrict__ deg) {
  int i = blockIdx.x * blockDim.x + threadIdx.x;
  if (i < NN) deg[i] = rsqrtf(deg[i]);  // deg >= 1 always
}

// ---------------------------------------------------------------------------
// f32 [K x N] (optionally batched) -> transposed f16 [N x K]
// ---------------------------------------------------------------------------
__global__ void cvt_t_f16_kernel(const float* __restrict__ src,
                                 _Float16* __restrict__ dst, int K, int N,
                                 int nmat) {
  int i = blockIdx.x * blockDim.x + threadIdx.x;
  int tot = K * N * nmat;
  if (i >= tot) return;
  int m = i / (K * N);
  int r = i - m * (K * N);
  int k = r / N;
  int n = r - k * N;
  dst[(size_t)m * K * N + (size_t)n * K + k] = (_Float16)src[i];
}

// ---------------------------------------------------------------------------
// Staging helper: 16 x C f32 rows -> f16 LDS tile (stride STR), packed b32
// ---------------------------------------------------------------------------
template <int C>
__device__ __forceinline__ void stage_tile(const float* __restrict__ src,
                                           _Float16* As, int row0, int tid) {
  for (int i = tid; i < 16 * (C / 2); i += 128) {
    int r = i / (C / 2);
    int c2 = 2 * (i - r * (C / 2));
    float2 v = *(const float2*)(src + (size_t)(row0 + r) * C + c2);
    union { _Float16 h[2]; unsigned u; } pk;
    pk.h[0] = (_Float16)v.x;
    pk.h[1] = (_Float16)v.y;
    *(unsigned*)(As + r * STR + c2) = pk.u;
  }
}

// ---------------------------------------------------------------------------
// lin1: h = x @ W (64x128) + b ; x0 = 0.5*h  (WMMA)
// ---------------------------------------------------------------------------
__global__ __launch_bounds__(128) void lin1_kernel(
    const float* __restrict__ x, const _Float16* __restrict__ Wt,
    const float* __restrict__ bias, float* __restrict__ h,
    float* __restrict__ x0) {
  __shared__ __align__(16) _Float16 Xs[16 * STR];
  const int row0 = blockIdx.x * 16;
  const int tid  = threadIdx.x;
  stage_tile<DIN>(x, Xs, row0, tid);
  __syncthreads();
  const int wave = tid >> 5, lane = tid & 31;
  const int hi = lane >> 4, nl = lane & 15;
#pragma unroll
  for (int t = 0; t < 2; ++t) {
    const int n0 = (wave + t * 4) * 16;
    v8f acc = {};
#pragma unroll
    for (int kk = 0; kk < DIN; kk += 32) {
      v16h a = frag_a_lds(Xs, lane, kk);
      v16h b = frag_b_glbT(Wt, DIN, lane, kk, n0);
      acc = __builtin_amdgcn_wmma_f32_16x16x32_f16(false, a, false, b,
                                                   (short)0, acc, false, false);
    }
    const int n = n0 + nl;
#pragma unroll
    for (int r = 0; r < 8; ++r) {
      int m = row0 + r + (hi ? 8 : 0);
      float v = acc[r] + bias[n];
      h[(size_t)m * DHID + n]  = v;
      x0[(size_t)m * DHID + n] = 0.5f * v;  // ALPHA * h0
    }
  }
}

// ---------------------------------------------------------------------------
// SpMM: agg = 0.5 * sum_{edges+loops} norm * h[row]
// ---------------------------------------------------------------------------
__global__ void spmm_init_kernel(const float* __restrict__ h,
                                 const float* __restrict__ dinv,
                                 float* __restrict__ agg) {
  int t = blockIdx.x * blockDim.x + threadIdx.x;
  if (t >= NN * (DHID / 4)) return;
  int node = t >> 5;
  int f = (t & 31) * 4;
  float w = 0.5f * dinv[node] * dinv[node];  // self-loop term * (1-ALPHA)
  float4 hv = *(const float4*)(h + (size_t)node * DHID + f);
  float4 o = {w * hv.x, w * hv.y, w * hv.z, w * hv.w};
  *(float4*)(agg + (size_t)node * DHID + f) = o;
}

__global__ void spmm_edge_kernel(const int* __restrict__ rows,
                                 const int* __restrict__ cols,
                                 const float* __restrict__ dinv,
                                 const float* __restrict__ h,
                                 float* __restrict__ agg) {
  int t = blockIdx.x * blockDim.x + threadIdx.x;  // NE*32 threads
  int e = t >> 5;
  if (e >= NE) return;
  int f = (t & 31) * 4;
  int r = rows[e], c = cols[e];
  float w = 0.5f * dinv[r] * dinv[c];
  float4 hv = *(const float4*)(h + (size_t)r * DHID + f);
  float* dst = agg + (size_t)c * DHID + f;
  atomicAdd(dst + 0, w * hv.x);
  atomicAdd(dst + 1, w * hv.y);
  atomicAdd(dst + 2, w * hv.z);
  atomicAdd(dst + 3, w * hv.w);
}

// ---------------------------------------------------------------------------
// Layer core: out = beta*(agg@W1 + x0@W2) + (1-beta)*(agg + x0)   (WMMA x2)
// ---------------------------------------------------------------------------
__global__ __launch_bounds__(128) void layer_gemm_kernel(
    const float* __restrict__ agg, const float* __restrict__ x0,
    const _Float16* __restrict__ W1t, const _Float16* __restrict__ W2t,
    float beta, float* __restrict__ out) {
  __shared__ __align__(16) _Float16 As[16 * STR];
  __shared__ __align__(16) _Float16 Xs[16 * STR];
  const int row0 = blockIdx.x * 16;
  const int tid  = threadIdx.x;
  stage_tile<DHID>(agg, As, row0, tid);
  stage_tile<DHID>(x0, Xs, row0, tid);
  __syncthreads();
  const int wave = tid >> 5, lane = tid & 31;
  const int hi = lane >> 4, nl = lane & 15;
  const float omb = 1.0f - beta;
#pragma unroll
  for (int t = 0; t < 2; ++t) {
    const int n0 = (wave + t * 4) * 16;
    v8f acc = {};
#pragma unroll
    for (int kk = 0; kk < DHID; kk += 32) {
      v16h a1 = frag_a_lds(As, lane, kk);
      v16h b1 = frag_b_glbT(W1t, DHID, lane, kk, n0);
      acc = __builtin_amdgcn_wmma_f32_16x16x32_f16(false, a1, false, b1,
                                                   (short)0, acc, false, false);
      v16h a2 = frag_a_lds(Xs, lane, kk);
      v16h b2 = frag_b_glbT(W2t, DHID, lane, kk, n0);
      acc = __builtin_amdgcn_wmma_f32_16x16x32_f16(false, a2, false, b2,
                                                   (short)0, acc, false, false);
    }
    const int n = n0 + nl;
#pragma unroll
    for (int r = 0; r < 8; ++r) {
      int m = row0 + r + (hi ? 8 : 0);
      size_t idx = (size_t)m * DHID + n;
      float ew = agg[idx] + x0[idx];
      out[idx] = beta * acc[r] + omb * ew;
    }
  }
}

// ---------------------------------------------------------------------------
// Graph-mode LayerNorm: global mean/std over all NN*DHID elements
// ---------------------------------------------------------------------------
__global__ __launch_bounds__(256) void reduce_kernel(
    const float* __restrict__ v, double* __restrict__ red) {
  double s = 0.0, s2 = 0.0;
  const int total = NN * DHID;
  for (int i = blockIdx.x * blockDim.x + threadIdx.x; i < total;
       i += gridDim.x * blockDim.x) {
    float f = v[i];
    s += (double)f;
    s2 += (double)f * (double)f;
  }
#pragma unroll
  for (int o = 16; o > 0; o >>= 1) {
    s  += __shfl_down(s, o, 32);
    s2 += __shfl_down(s2, o, 32);
  }
  __shared__ double sh[8], sh2[8];
  int lane = threadIdx.x & 31, w = threadIdx.x >> 5;
  if (lane == 0) { sh[w] = s; sh2[w] = s2; }
  __syncthreads();
  if (threadIdx.x == 0) {
    double a = 0.0, b = 0.0;
#pragma unroll
    for (int i = 0; i < 8; ++i) { a += sh[i]; b += sh2[i]; }
    atomicAdd(&red[0], a);
    atomicAdd(&red[1], b);
  }
}

__global__ void norm_relu_kernel(const float* __restrict__ out,
                                 const double* __restrict__ red,
                                 const float* __restrict__ nw,
                                 const float* __restrict__ nb,
                                 float* __restrict__ h) {
  int t = blockIdx.x * blockDim.x + threadIdx.x;
  if (t >= NN * (DHID / 4)) return;
  const double M = (double)NN * (double)DHID;
  double mean = red[0] / M;
  double var = red[1] / M - mean * mean;
  float sd = (float)sqrt(var > 0.0 ? var : 0.0);
  float inv = 1.0f / (sd + LN_EPS);
  float mu = (float)mean;
  int node = t >> 5;
  int f = (t & 31) * 4;
  float4 ov = *(const float4*)(out + (size_t)node * DHID + f);
  float4 r;
  r.x = fmaxf((ov.x - mu) * inv * nw[f + 0] + nb[f + 0], 0.0f);
  r.y = fmaxf((ov.y - mu) * inv * nw[f + 1] + nb[f + 1], 0.0f);
  r.z = fmaxf((ov.z - mu) * inv * nw[f + 2] + nb[f + 2], 0.0f);
  r.w = fmaxf((ov.w - mu) * inv * nw[f + 3] + nb[f + 3], 0.0f);
  *(float4*)(h + (size_t)node * DHID + f) = r;
}

// ---------------------------------------------------------------------------
// lin2: out = h @ W (128x64) + b   (WMMA)
// ---------------------------------------------------------------------------
__global__ __launch_bounds__(128) void lin2_kernel(
    const float* __restrict__ h, const _Float16* __restrict__ Wt,
    const float* __restrict__ bias, float* __restrict__ out) {
  __shared__ __align__(16) _Float16 Hs[16 * STR];
  const int row0 = blockIdx.x * 16;
  const int tid  = threadIdx.x;
  stage_tile<DHID>(h, Hs, row0, tid);
  __syncthreads();
  const int wave = tid >> 5, lane = tid & 31;
  const int hi = lane >> 4, nl = lane & 15;
  const int n0 = wave * 16;  // DOUT/16 == 4 tiles, one per wave
  v8f acc = {};
#pragma unroll
  for (int kk = 0; kk < DHID; kk += 32) {
    v16h a = frag_a_lds(Hs, lane, kk);
    v16h b = frag_b_glbT(Wt, DHID, lane, kk, n0);
    acc = __builtin_amdgcn_wmma_f32_16x16x32_f16(false, a, false, b, (short)0,
                                                 acc, false, false);
  }
  const int n = n0 + nl;
#pragma unroll
  for (int r = 0; r < 8; ++r) {
    int m = row0 + r + (hi ? 8 : 0);
    out[(size_t)m * DOUT + n] = acc[r] + bias[n];
  }
}

// ---------------------------------------------------------------------------
// Host-side orchestration
// ---------------------------------------------------------------------------
extern "C" void kernel_launch(void* const* d_in, const int* in_sizes, int n_in,
                              void* d_out, int out_size, void* d_ws,
                              size_t ws_size, hipStream_t stream) {
  (void)in_sizes; (void)n_in; (void)out_size; (void)ws_size;
  const float* x      = (const float*)d_in[0];
  const int*   ei     = (const int*)d_in[1];  // (2, NE): rows then cols
  const float* lin1_w = (const float*)d_in[2];
  const float* lin1_b = (const float*)d_in[3];
  const float* w1     = (const float*)d_in[4];  // (8,128,128)
  const float* w2     = (const float*)d_in[5];
  const float* norm_w = (const float*)d_in[6];
  const float* norm_b = (const float*)d_in[7];
  const float* lin2_w = (const float*)d_in[8];
  const float* lin2_b = (const float*)d_in[9];
  float* out_final = (float*)d_out;

  char* ws = (char*)d_ws;
  size_t off = 0;
  auto alloc = [&](size_t bytes) -> char* {
    char* p = ws + off;
    off = (off + bytes + 255) & ~(size_t)255;
    return p;
  };
  float* dinv = (float*)alloc((size_t)NN * 4);
  float* h    = (float*)alloc((size_t)NN * DHID * 4);
  float* x0   = (float*)alloc((size_t)NN * DHID * 4);
  float* agg  = (float*)alloc((size_t)NN * DHID * 4);
  float* tmp  = (float*)alloc((size_t)NN * DHID * 4);
  _Float16* lin1t = (_Float16*)alloc((size_t)DIN * DHID * 2);
  _Float16* w1t   = (_Float16*)alloc((size_t)NLAYERS * DHID * DHID * 2);
  _Float16* w2t   = (_Float16*)alloc((size_t)NLAYERS * DHID * DHID * 2);
  _Float16* lin2t = (_Float16*)alloc((size_t)DHID * DOUT * 2);
  double* red = (double*)alloc(16);

  const int T = 256;
  // gcn_norm: degree (col-segments, self-loops folded in) -> rsqrt
  deg_init_kernel<<<(NN + T - 1) / T, T, 0, stream>>>(dinv);
  deg_count_kernel<<<(NE + T - 1) / T, T, 0, stream>>>(ei + NE, dinv);
  deg_to_dinv_kernel<<<(NN + T - 1) / T, T, 0, stream>>>(dinv);

  // weights -> transposed f16 (fragment-friendly, once per launch)
  cvt_t_f16_kernel<<<(DIN * DHID + T - 1) / T, T, 0, stream>>>(lin1_w, lin1t,
                                                               DIN, DHID, 1);
  cvt_t_f16_kernel<<<(NLAYERS * DHID * DHID + T - 1) / T, T, 0, stream>>>(
      w1, w1t, DHID, DHID, NLAYERS);
  cvt_t_f16_kernel<<<(NLAYERS * DHID * DHID + T - 1) / T, T, 0, stream>>>(
      w2, w2t, DHID, DHID, NLAYERS);
  cvt_t_f16_kernel<<<(DHID * DOUT + T - 1) / T, T, 0, stream>>>(lin2_w, lin2t,
                                                                DHID, DOUT, 1);

  // lin1 + x0 = ALPHA*h0
  lin1_kernel<<<NN / 16, 128, 0, stream>>>(x, lin1t, lin1_b, h, x0);

  for (int i = 0; i < NLAYERS; ++i) {
    float beta = logf(1.0f / (float)(i + 1) + 1.0f);
    spmm_init_kernel<<<(NN * 32 + T - 1) / T, T, 0, stream>>>(h, dinv, agg);
    spmm_edge_kernel<<<(NE * 32) / T, T, 0, stream>>>(ei, ei + NE, dinv, h,
                                                      agg);
    layer_gemm_kernel<<<NN / 16, 128, 0, stream>>>(
        agg, x0, w1t + (size_t)i * DHID * DHID, w2t + (size_t)i * DHID * DHID,
        beta, tmp);
    hipMemsetAsync(red, 0, 16, stream);
    reduce_kernel<<<1024, 256, 0, stream>>>(tmp, red);
    norm_relu_kernel<<<(NN * 32 + T - 1) / T, T, 0, stream>>>(
        tmp, red, norm_w + i * DHID, norm_b + i * DHID, h);
  }

  lin2_kernel<<<NN / 16, 128, 0, stream>>>(h, lin2t, lin2_b, out_final);
}
